// MultiHeadAttention_15032385536440
// MI455X (gfx1250) — matrix operand order repaired and checked by reading, not compiled, not run on hardware
//
#include <hip/hip_runtime.h>
#include <hip/hip_bf16.h>
#include <cstdint>

// ---------------------------------------------------------------------------
// MHA (B=4,T=2048,E=1024,H=16,D=64) for gfx1250:
//   kernel 1: x -> Q,K,V bf16 in d_ws (+ transposed Vt[q][t]) via bf16 WMMA
//   kernel 2: flash attention, K tiles staged to LDS by the Tensor Data Mover
//             (double buffered, s_wait_tensorcnt), V fragments streamed from
//             L2-resident Vt, online softmax, + V residual.
// ---------------------------------------------------------------------------

typedef __attribute__((ext_vector_type(16))) __bf16 v16bf;
typedef __attribute__((ext_vector_type(8)))  __bf16 v8bf;
typedef __attribute__((ext_vector_type(8)))  float  v8f;
typedef __attribute__((ext_vector_type(4)))  unsigned int u32x4;
typedef __attribute__((ext_vector_type(8)))  int i32x8;
typedef __attribute__((ext_vector_type(4)))  int i32x4;

#define B_ 4
#define T_ 2048
#define E_ 1024
#define H_ 16
#define D_ 64

#if defined(__gfx1250__) && __has_builtin(__builtin_amdgcn_tensor_load_to_lds)
#define HAS_TDM 1
#else
#define HAS_TDM 0
#endif

__device__ __forceinline__ v16bf cat8(v8bf lo, v8bf hi) {
  v16bf r;
#pragma unroll
  for (int i = 0; i < 8; ++i) { r[i] = lo[i]; r[i + 8] = hi[i]; }
  return r;
}

#if HAS_TDM
// Issue a TDM load of a 32x64 bf16 tile (contiguous rows, row stride 64) into
// LDS. D# per CDNA5 ISA. Tracked by TENSORcnt.
// This toolchain's builtin takes 6 args (4 SGPR groups + spare group + cpol).
__device__ __forceinline__ void tdm_load_k_tile(const __bf16* gsrc, void* lds_dst) {
  const unsigned long long ga = (unsigned long long)(uintptr_t)gsrc;
  const unsigned lds = (unsigned)(uintptr_t)lds_dst;  // low 32 bits = LDS offset
  u32x4 g0;
  g0[0] = 1u;                                          // count=1 (valid), user mode
  g0[1] = lds;                                         // lds_addr
  g0[2] = (unsigned)(ga & 0xFFFFFFFFu);                // global_addr[31:0]
  g0[3] = (unsigned)((ga >> 32) & 0x01FFFFFFu)         // global_addr[56:32]
          | (2u << 30);                                // type=2 ("image")
  i32x8 g1;
  g1[0] = (1 << 16);                                   // data_size=1 -> 2 bytes
  g1[1] = (64 << 16);                                  // tensor_dim0 = 64
  g1[2] = (int)(2048u << 16);                          // tensor_dim1 = 2048
  g1[3] = (64 << 16);                                  // tile_dim0 = 64
  g1[4] = 32;                                          // tile_dim1 = 32
  g1[5] = 64;                                          // tensor_dim0_stride = 64
  g1[6] = 0;
  g1[7] = 0;
  const i32x4 z4 = {0, 0, 0, 0};                       // 2-D tile: groups 2/3 unused
  const i32x8 z8 = {0, 0, 0, 0, 0, 0, 0, 0};
  __builtin_amdgcn_tensor_load_to_lds(g0, g1, z4, z4, z8, 0);
}
#endif

__device__ __forceinline__ void wait_tensor0() {
#if HAS_TDM
#if __has_builtin(__builtin_amdgcn_s_wait_tensorcnt)
  __builtin_amdgcn_s_wait_tensorcnt(0);
#else
  asm volatile("s_wait_tensorcnt 0" ::: "memory");
#endif
  asm volatile("" ::: "memory");   // keep LDS reads below the wait
#endif
}

// ---------------------------------------------------------------------------
// Kernel 1: QKV projection. One wave = one 16-row tile of xh[b,h,:,:].
// Also emits transposed Vt[bh][q][t] so kernel 2 gets contiguous V B-frags.
// ---------------------------------------------------------------------------
__global__ __launch_bounds__(128)
void qkv_proj_kernel(const float* __restrict__ x,
                     const float* __restrict__ Wq,
                     const float* __restrict__ Wk,
                     const float* __restrict__ Wv,
                     __bf16* __restrict__ Qb,
                     __bf16* __restrict__ Kb,
                     __bf16* __restrict__ Vb,
                     __bf16* __restrict__ Vt) {
  __shared__ __align__(32) __bf16 Ws[3][64][64];   // [w][q][d], 24 KB

  const int tid = threadIdx.x;
  for (int i = tid; i < 64 * 64; i += 128) {
    (&Ws[0][0][0])[i] = (__bf16)Wq[i];
    (&Ws[1][0][0])[i] = (__bf16)Wk[i];
    (&Ws[2][0][0])[i] = (__bf16)Wv[i];
  }
  __syncthreads();

  const int wave = tid >> 5;
  const int lane = tid & 31;
  const int g    = lane >> 4;
  const int mn   = lane & 15;

  const int tile = blockIdx.x * 4 + wave;   // 0..8191
  const int bh   = tile >> 7;
  const int t0   = (tile & 127) << 4;
  const int b    = bh / H_;
  const int h    = bh % H_;

  const float* xrow = x + (size_t)(b * T_ + t0 + mn) * E_ + h * D_;
  v16bf a[2];
#pragma unroll
  for (int ks = 0; ks < 2; ++ks) {
    const int d0 = ks * 32 + 16 * g;
    v8f flo = *(const v8f*)(xrow + d0);
    v8f fhi = *(const v8f*)(xrow + d0 + 8);
    v8bf lo, hi;
#pragma unroll
    for (int i = 0; i < 8; ++i) { lo[i] = (__bf16)flo[i]; hi[i] = (__bf16)fhi[i]; }
    a[ks] = cat8(lo, hi);
  }

  __bf16* outp[3] = {Qb, Kb, Vb};
#pragma unroll
  for (int w = 0; w < 3; ++w) {
#pragma unroll
    for (int qb = 0; qb < 4; ++qb) {
      v8f c = {};
#pragma unroll
      for (int ks = 0; ks < 2; ++ks) {
        const v16bf bfrag = *(const v16bf*)&Ws[w][qb * 16 + mn][ks * 32 + 16 * g];
        c = __builtin_amdgcn_wmma_f32_16x16x32_bf16(
                false, a[ks], false, bfrag, (short)0, c, false, false);
      }
      __bf16* op = outp[w] + (size_t)(bh * T_ + t0) * D_ + qb * 16 + mn;
#pragma unroll
      for (int r = 0; r < 8; ++r)
        op[(size_t)(r + 8 * g) * D_] = (__bf16)c[r];
      if (w == 2) {  // transposed V: rows contiguous in t -> merged b128 stores
        __bf16* tp = Vt + ((size_t)bh * 64 + qb * 16 + mn) * T_ + t0 + 8 * g;
#pragma unroll
        for (int r = 0; r < 8; ++r) tp[r] = (__bf16)c[r];
      }
    }
  }
}

// ---------------------------------------------------------------------------
// Kernel 2: flash attention. Block = 4 waves = 64 Q rows of one (b,h).
// K tiles: TDM double-buffered into LDS (fallback: cooperative copy).
// V B-frags: contiguous 32B global loads from Vt (L2 resident).
// ---------------------------------------------------------------------------
__global__ __launch_bounds__(128)
void flash_attn_kernel(const __bf16* __restrict__ Qb,
                       const __bf16* __restrict__ Kb,
                       const __bf16* __restrict__ Vb,
                       const __bf16* __restrict__ Vt,
                       float* __restrict__ out) {
  __shared__ __align__(32) __bf16 Ks[2][32][64];   // double-buffered K tile, 8 KB
  __shared__ __align__(32) __bf16 Ps[4][16][32];   // per-wave P tile,       4 KB

  const int tid  = threadIdx.x;
  const int wave = tid >> 5;
  const int lane = tid & 31;
  const int g    = lane >> 4;
  const int mn   = lane & 15;

  const int blk = blockIdx.x;           // B*H*(T/64) = 2048 blocks
  const int bh  = blk >> 5;
  const int t0  = ((blk & 31) << 6) + wave * 16;

  // Q A-fragments for this wave's 16 rows
  v16bf aq[2];
  {
    const __bf16* qrow = Qb + (size_t)(bh * T_ + t0 + mn) * D_;
#pragma unroll
    for (int ks = 0; ks < 2; ++ks) {
      const int d0 = ks * 32 + 16 * g;
      v8bf lo = *(const v8bf*)(qrow + d0);
      v8bf hi = *(const v8bf*)(qrow + d0 + 8);
      aq[ks] = cat8(lo, hi);
    }
  }

  v8f o[4] = {};
  float mrow[8], lrow[8];
#pragma unroll
  for (int r = 0; r < 8; ++r) { mrow[r] = -3.0e38f; lrow[r] = 0.0f; }

  const __bf16* Kbase  = Kb + (size_t)bh * T_ * D_;
  const __bf16* Vtbase = Vt + (size_t)bh * 64 * T_;   // [q][t]
  const float scale = 0.125f;   // 1/sqrt(64)

#if HAS_TDM
  tdm_load_k_tile(Kbase, &Ks[0][0][0]);   // prologue: tile 0 -> buffer 0
#endif

  for (int s0 = 0; s0 < T_; s0 += 32) {
    const int buf = (s0 >> 5) & 1;
    __syncthreads();   // all waves done reading buf^1 before it is refilled

#if HAS_TDM
    wait_tensor0();    // tile s0 resident in Ks[buf]
    if (s0 + 32 < T_)  // overlap next tile's DMA with this tile's WMMAs
      tdm_load_k_tile(Kbase + (size_t)(s0 + 32) * D_, &Ks[buf ^ 1][0][0]);
#else
    {
      const uint4* src = (const uint4*)(Kbase + (size_t)s0 * D_);
      uint4* dst = (uint4*)&Ks[buf][0][0];
      for (int i = tid; i < 256; i += 128) dst[i] = src[i];
    }
    __syncthreads();
#endif
    if (s0 + 32 < T_)   // warm next Vt stripe (global_prefetch_b8)
      __builtin_prefetch(Vtbase + (size_t)(tid & 63) * T_ + s0 + 32, 0, 1);

    // S = Q K^T : two 16-col blocks, K=64 in two k-steps
    v8f s[2];
#pragma unroll
    for (int nb = 0; nb < 2; ++nb) {
      v8f c = {};
#pragma unroll
      for (int ks = 0; ks < 2; ++ks) {
        const v16bf bk = *(const v16bf*)&Ks[buf][nb * 16 + mn][ks * 32 + 16 * g];
        c = __builtin_amdgcn_wmma_f32_16x16x32_bf16(
                false, aq[ks], false, bk, (short)0, c, false, false);
      }
      s[nb] = c;
    }

    // online softmax over the 32 new columns
#pragma unroll
    for (int r = 0; r < 8; ++r) {
      const float v0 = s[0][r] * scale;
      const float v1 = s[1][r] * scale;
      float mx = fmaxf(v0, v1);
      mx = fmaxf(mx, __shfl_xor(mx, 1, 32));
      mx = fmaxf(mx, __shfl_xor(mx, 2, 32));
      mx = fmaxf(mx, __shfl_xor(mx, 4, 32));
      mx = fmaxf(mx, __shfl_xor(mx, 8, 32));  // masks<16 keep lane-halves apart
      const float mnew  = fmaxf(mrow[r], mx);
      const float alpha = __expf(mrow[r] - mnew);
      const float p0 = __expf(v0 - mnew);
      const float p1 = __expf(v1 - mnew);
      float ps = p0 + p1;
      ps += __shfl_xor(ps, 1, 32);
      ps += __shfl_xor(ps, 2, 32);
      ps += __shfl_xor(ps, 4, 32);
      ps += __shfl_xor(ps, 8, 32);
      lrow[r] = lrow[r] * alpha + ps;
      mrow[r] = mnew;
#pragma unroll
      for (int qb = 0; qb < 4; ++qb) o[qb][r] *= alpha;
      Ps[wave][r + 8 * g][mn]      = (__bf16)p0;   // C layout: row r+8g
      Ps[wave][r + 8 * g][16 + mn] = (__bf16)p1;
    }
    // wave-private LDS RAW: DS ops in-order per wave; fence the compiler
    asm volatile("s_wait_dscnt 0" ::: "memory");

    // reload P as A fragment (16x32)
    const __bf16* prow = &Ps[wave][mn][0];
    const v8bf plo = *(const v8bf*)(prow + 16 * g);
    const v8bf phi = *(const v8bf*)(prow + 16 * g + 8);
    const v16bf pa = cat8(plo, phi);

    // O += P * V   (B[k=s, n=q] = Vt[q][s0+s], contiguous 32B per lane)
#pragma unroll
    for (int qb = 0; qb < 4; ++qb) {
      const v16bf bv =
          *(const v16bf*)(Vtbase + (size_t)(qb * 16 + mn) * T_ + s0 + 16 * g);
      o[qb] = __builtin_amdgcn_wmma_f32_16x16x32_bf16(
                  false, pa, false, bv, (short)0, o[qb], false, false);
    }
  }

  // epilogue: normalize, add V residual, scatter to [B,T,E] f32
  const int b = bh / H_, h = bh % H_;
#pragma unroll
  for (int qb = 0; qb < 4; ++qb) {
#pragma unroll
    for (int r = 0; r < 8; ++r) {
      const int t = t0 + r + 8 * g;
      const int q = qb * 16 + mn;
      const float vres = (float)Vb[(size_t)(bh * T_ + t) * D_ + q];
      out[(size_t)(b * T_ + t) * E_ + h * D_ + q] = o[qb][r] / lrow[r] + vres;
    }
  }
}

// ---------------------------------------------------------------------------
extern "C" void kernel_launch(void* const* d_in, const int* in_sizes, int n_in,
                              void* d_out, int out_size, void* d_ws, size_t ws_size,
                              hipStream_t stream) {
  const float* x  = (const float*)d_in[0];
  const float* Wq = (const float*)d_in[1];
  const float* Wk = (const float*)d_in[2];
  const float* Wv = (const float*)d_in[3];

  const size_t elems = (size_t)B_ * H_ * T_ * D_;   // 8,388,608 per tensor
  __bf16* Qb = (__bf16*)d_ws;
  __bf16* Kb = Qb + elems;
  __bf16* Vb = Kb + elems;
  __bf16* Vt = Vb + elems;                          // transposed V [bh][q][t]

  qkv_proj_kernel<<<2048, 128, 0, stream>>>(x, Wq, Wk, Wv, Qb, Kb, Vb, Vt);
  flash_attn_kernel<<<2048, 128, 0, stream>>>(Qb, Kb, Vb, Vt, (float*)d_out);
}
